// TransformerEncoderLayer_89086211653704
// MI455X (gfx1250) — matrix operand order repaired and checked by reading, not compiled, not run on hardware
//
#include <hip/hip_runtime.h>
#include <hip/hip_bf16.h>
#include <stdint.h>

typedef __attribute__((ext_vector_type(16))) __bf16 v16bf;
typedef __attribute__((ext_vector_type(8)))  float  v8f;

using u16 = unsigned short;
using u32 = unsigned int;

#define DEV __device__ __forceinline__

union Frag { v16bf v; uint4 q[2]; u16 u[16]; };

DEV u16 f2bf(float f) {              // round-to-nearest-even f32 -> bf16
  u32 x = __float_as_uint(f);
  x += 0x7fffu + ((x >> 16) & 1u);
  return (u16)(x >> 16);
}
DEV float bf2f(u16 h) { return __uint_as_float(((u32)h) << 16); }

// CDNA5 async global->LDS copy (ASYNCcnt path, no VGPR landing).
// lds_off: byte offset of destination within LDS (low 32 bits of generic ptr).
DEV void async_g2l_b128(unsigned lds_off, const void* gptr) {
  asm volatile("global_load_async_to_lds_b128 %0, %1, off"
               :: "v"(lds_off), "v"((unsigned long long)gptr)
               : "memory");
}
DEV void wait_async0() { asm volatile("s_wait_asynccnt 0" ::: "memory"); }
DEV unsigned lds_off32(const void* p) { return (unsigned)(size_t)p; }

// ---------------------------------------------------------------------------
// Generic bf16 WMMA GEMM: C[M,N] = A[M,K] * B[K,N] (+bias)
// Block tile 128x128, BK=32, 256 threads = 8 waves (2x4), wave tile 64x32.
// A tile staged via GLOBAL_LOAD_ASYNC_TO_LDS_B128; B tile load+transpose.
// EPI: 0 = f32 out, 1 = bf16 out, 2 = QKV scatter (bias + per-head layout)
// ---------------------------------------------------------------------------
template<int EPI>
__launch_bounds__(256, 2)
__global__ void gemm_bf16(const u16* __restrict__ A, const u16* __restrict__ Bw,
                          const float* __restrict__ bias,
                          float* __restrict__ Cf, u16* __restrict__ Cb,
                          u16* __restrict__ Qp, u16* __restrict__ Kp, u16* __restrict__ Vp,
                          int M, int N, int K)
{
  constexpr int BM = 128, BN = 128, BK = 32, LDA = 40;  // 40 halves = 80B, 16B aligned
  __shared__ __align__(16) u16 As[BM * LDA];
  __shared__ __align__(16) u16 Bs[BN * LDA];            // transposed: [n][k]

  const int tid = threadIdx.x, lane = tid & 31, wid = tid >> 5;
  const int half = lane >> 4, l16 = lane & 15;
  const int bx = blockIdx.x % (N / BN), by = blockIdx.x / (N / BN);
  const int wm = wid >> 2, wn = wid & 3;                // waves: 2 (M) x 4 (N)

  v8f acc[4][2] = {};
  const int kTiles = K / BK;

  for (int kt = 0; kt < kTiles; ++kt) {
    __syncthreads();
    // Stage A tile 128x32 (row-major, k contiguous) via async DMA to LDS
#pragma unroll
    for (int i = 0; i < 2; ++i) {
      int c = tid + i * 256;                 // 512 chunks of 8 halves
      int row = c >> 2, kc = c & 3;
      async_g2l_b128(lds_off32(As + row * LDA + kc * 8),
                     A + (size_t)(by * BM + row) * K + kt * BK + kc * 8);
    }
    // Stage B tile 32x128 transposed into Bs[n][k] (manual: needs transpose)
#pragma unroll
    for (int i = 0; i < 2; ++i) {
      int c = tid + i * 256;
      int kr = c >> 4, nc = c & 15;
      union { uint4 q; u16 u[8]; } t;
      t.q = *(const uint4*)(Bw + (size_t)(kt * BK + kr) * N + bx * BN + nc * 8);
#pragma unroll
      for (int j = 0; j < 8; ++j) Bs[(nc * 8 + j) * LDA + kr] = t.u[j];
    }
    if (kt + 1 < kTiles) {                   // CDNA5 global_prefetch_b8 of next tiles
      __builtin_prefetch(A + (size_t)(by * BM + (tid >> 1)) * K + (kt + 1) * BK, 0, 1);
      __builtin_prefetch(Bw + (size_t)((kt + 1) * BK + (tid & 31)) * N + bx * BN, 0, 1);
    }
    wait_async0();                           // async LDS writes visible
    __syncthreads();

    Frag a[4], b[2];
#pragma unroll
    for (int mt = 0; mt < 4; ++mt) {         // A frag: lane m=l16, K = {half*8..}+{16+half*8..}
      const u16* p = As + (wm * 64 + mt * 16 + l16) * LDA;
      a[mt].q[0] = *(const uint4*)(p + half * 8);
      a[mt].q[1] = *(const uint4*)(p + 16 + half * 8);
    }
#pragma unroll
    for (int nt = 0; nt < 2; ++nt) {         // B frag: lane n=l16, K = {half*8..}+{16+half*8..}
      const u16* p = Bs + (wn * 32 + nt * 16 + l16) * LDA;
      b[nt].q[0] = *(const uint4*)(p + half * 8);
      b[nt].q[1] = *(const uint4*)(p + 16 + half * 8);
    }
#pragma unroll
    for (int mt = 0; mt < 4; ++mt)
#pragma unroll
      for (int nt = 0; nt < 2; ++nt)
        acc[mt][nt] = __builtin_amdgcn_wmma_f32_16x16x32_bf16(
            false, a[mt].v, false, b[nt].v, (short)0, acc[mt][nt], false, false);
  }

  // Epilogue: C layout row = r + 8*half, col = l16 within 16x16 tile
#pragma unroll
  for (int mt = 0; mt < 4; ++mt)
#pragma unroll
    for (int nt = 0; nt < 2; ++nt) {
      int col = bx * BN + wn * 32 + nt * 16 + l16;
      float bv = bias ? bias[col] : 0.f;
#pragma unroll
      for (int r = 0; r < 8; ++r) {
        int row = by * BM + wm * 64 + mt * 16 + r + half * 8;
        float v = acc[mt][nt][r] + bv;
        if constexpr (EPI == 0) {
          Cf[(size_t)row * N + col] = v;
        } else if constexpr (EPI == 1) {
          Cb[(size_t)row * N + col] = f2bf(v);
        } else {                              // QKV scatter: row = b*2048+s
          int b_ = row >> 11, s = row & 2047;
          if (col < 1024) {
            int h = col >> 6, d = col & 63;
            Qp[(((size_t)(b_ * 16 + h)) * 2048 + s) * 64 + d] = f2bf(v);
          } else if (col < 1280) {
            int kv = (col - 1024) >> 6, d = col & 63;
            Kp[(((size_t)(b_ * 4 + kv)) * 2048 + s) * 64 + d] = f2bf(v);
          } else {
            int kv = (col - 1280) >> 6, d = col & 63;
            Vp[(((size_t)(b_ * 4 + kv)) * 2048 + s) * 64 + d] = f2bf(v);
          }
        }
      }
    }
}

// ---------------------------------------------------------------------------
// Flash attention, GQA (16 Q heads / 4 KV heads), ALiBi, key mask.
// 128 threads = 4 waves; block handles 64 q-rows of one (batch, head).
// K tile staged via async DMA; V tile load+transpose; scores & P*V via WMMA.
// ---------------------------------------------------------------------------
__launch_bounds__(128, 2)
__global__ void attention_kernel(const u16* __restrict__ Qb, const u16* __restrict__ Kb,
                                 const u16* __restrict__ Vb, const int* __restrict__ amask,
                                 u16* __restrict__ ctx)
{
  constexpr int S = 2048, DK = 64;
  __shared__ __align__(16) u16 Kt[32 * 72];      // [key][d], 144B row stride (16B mult)
  __shared__ __align__(16) u16 Vt[64 * 40];      // [d][key] transposed
  __shared__ __align__(16) u16 Pw[4 * 16 * 40];  // per-wave P tiles [16][32]+pad

  const int tid = threadIdx.x, lane = tid & 31, wid = tid >> 5;
  const int half = lane >> 4, l16 = lane & 15;
  const int qt = blockIdx.x, h = blockIdx.y, bz = blockIdx.z;
  const float slope = exp2f(-(float)(h + 1));    // ALIBI_ALPHA = 1

  // Q fragments for this wave's 16 q-rows (A layout; two K=32 steps over d)
  const u16* Qrow = Qb + (((size_t)(bz * 16 + h)) * S + qt * 64 + wid * 16 + l16) * DK;
  Frag qf[2];
#pragma unroll
  for (int c = 0; c < 2; ++c) {
    qf[c].q[0] = *(const uint4*)(Qrow + c * 32 + half * 8);
    qf[c].q[1] = *(const uint4*)(Qrow + c * 32 + 16 + half * 8);
  }
  const u16* Kh = Kb + ((size_t)(bz * 4 + (h >> 2))) * S * DK;
  const u16* Vh = Vb + ((size_t)(bz * 4 + (h >> 2))) * S * DK;

  v8f cacc[4] = {};
  float mrow[8], lrow[8];
#pragma unroll
  for (int r = 0; r < 8; ++r) { mrow[r] = -1e30f; lrow[r] = 0.f; }

  for (int kb = 0; kb < S / 32; ++kb) {
    __syncthreads();
    // Stage K (async DMA, row-major) and V (manual transpose): 32 keys x 64 d
#pragma unroll
    for (int i = 0; i < 2; ++i) {
      int c = tid + i * 128;                 // 256 chunks of 8 halves
      int key = c >> 3, dc = c & 7;
      async_g2l_b128(lds_off32(Kt + key * 72 + dc * 8),
                     Kh + (size_t)(kb * 32 + key) * DK + dc * 8);
      union { uint4 q; u16 u[8]; } vv;
      vv.q = *(const uint4*)(Vh + (size_t)(kb * 32 + key) * DK + dc * 8);
#pragma unroll
      for (int j = 0; j < 8; ++j) Vt[(dc * 8 + j) * 40 + key] = vv.u[j];
    }
    wait_async0();
    __syncthreads();

    // Scores: two 16x16 key tiles, each chained over d = 0..31, 32..63
    v8f sacc[2] = {};
#pragma unroll
    for (int t = 0; t < 2; ++t)
#pragma unroll
      for (int c = 0; c < 2; ++c) {
        Frag kf;
        const u16* p = Kt + (t * 16 + l16) * 72 + c * 32 + half * 8;
        kf.q[0] = *(const uint4*)(p);
        kf.q[1] = *(const uint4*)(p + 16);
        sacc[t] = __builtin_amdgcn_wmma_f32_16x16x32_bf16(
            false, qf[c].v, false, kf.v, (short)0, sacc[t], false, false);
      }

    const int k0 = kb * 32 + l16, k1 = k0 + 16;
    const float mb0 = (amask[bz * S + k0] == 0) ? -1e9f : 0.f;
    const float mb1 = (amask[bz * S + k1] == 0) ? -1e9f : 0.f;

    float s0[8], s1[8];
#pragma unroll
    for (int r = 0; r < 8; ++r) {
      float qi = (float)(qt * 64 + wid * 16 + r + half * 8);
      s0[r] = sacc[0][r] * 0.125f + slope * (qi - (float)k0) + mb0;
      s1[r] = sacc[1][r] * 0.125f + slope * (qi - (float)k1) + mb1;
    }
    // Online softmax; row lives in one 16-lane half -> xor-butterfly < 16
#pragma unroll
    for (int r = 0; r < 8; ++r) {
      float mx = fmaxf(s0[r], s1[r]);
#pragma unroll
      for (int off = 1; off < 16; off <<= 1) mx = fmaxf(mx, __shfl_xor(mx, off, 32));
      float mnew = fmaxf(mrow[r], mx);
      float scale = __expf(mrow[r] - mnew);
      float p0 = __expf(s0[r] - mnew);
      float p1 = __expf(s1[r] - mnew);
      float rs = p0 + p1;
#pragma unroll
      for (int off = 1; off < 16; off <<= 1) rs += __shfl_xor(rs, off, 32);
      lrow[r] = lrow[r] * scale + rs;
      mrow[r] = mnew;
#pragma unroll
      for (int j = 0; j < 4; ++j) cacc[j][r] *= scale;
      int prow = r + half * 8;
      Pw[(wid * 16 + prow) * 40 + l16]      = f2bf(p0);
      Pw[(wid * 16 + prow) * 40 + 16 + l16] = f2bf(p1);
    }
    asm volatile("s_wait_dscnt 0" ::: "memory");   // in-wave LDS transpose fence

    Frag pf;
    const u16* pp = Pw + (wid * 16 + l16) * 40;
    pf.q[0] = *(const uint4*)(pp + half * 8);
    pf.q[1] = *(const uint4*)(pp + 16 + half * 8);
#pragma unroll
    for (int j = 0; j < 4; ++j) {                  // ctx += P * V, 4 d-tiles
      Frag vf;
      const u16* vp = Vt + (j * 16 + l16) * 40 + half * 8;
      vf.q[0] = *(const uint4*)(vp);
      vf.q[1] = *(const uint4*)(vp + 16);
      cacc[j] = __builtin_amdgcn_wmma_f32_16x16x32_bf16(
          false, pf.v, false, vf.v, (short)0, cacc[j], false, false);
    }
    if (kb + 1 < S / 32) {
      __builtin_prefetch(Kh + (size_t)((kb + 1) * 32 + (tid & 31)) * DK, 0, 1);
      __builtin_prefetch(Vh + (size_t)((kb + 1) * 32 + (tid & 31)) * DK, 0, 1);
    }
  }

  // Finalize: ctx row-major [b*S+s][H*64] bf16 for the Wo GEMM
#pragma unroll
  for (int r = 0; r < 8; ++r) {
    float inv = 1.f / lrow[r];
    int qrow = qt * 64 + wid * 16 + r + half * 8;
#pragma unroll
    for (int j = 0; j < 4; ++j) {
      int col = h * 64 + j * 16 + l16;
      ctx[((size_t)(bz * S + qrow)) * 1024 + col] = f2bf(cacc[j][r] * inv);
    }
  }
}

// ---------------------------------------------------------------------------
// Residual add + LayerNorm: out = LN(X + maybe_masked(Y)); f32 (+ bf16) out
// ---------------------------------------------------------------------------
__launch_bounds__(256, 4)
__global__ void ln_kernel(const float* __restrict__ X, const float* __restrict__ Y,
                          const unsigned char* __restrict__ pad,
                          const float* __restrict__ g, const float* __restrict__ bta,
                          float* __restrict__ outf, u16* __restrict__ outb)
{
  constexpr int D = 1024;
  __shared__ float red[256];
  const int row = blockIdx.x, tid = threadIdx.x;
  const bool pz = pad && pad[row];

  float v[4], s = 0.f;
#pragma unroll
  for (int i = 0; i < 4; ++i) {
    int c = tid + i * 256;
    float y = Y[(size_t)row * D + c];
    if (pz) y = 0.f;
    v[i] = X[(size_t)row * D + c] + y;
    s += v[i];
  }
  red[tid] = s; __syncthreads();
  for (int o = 128; o > 0; o >>= 1) { if (tid < o) red[tid] += red[tid + o]; __syncthreads(); }
  float mu = red[0] / D;
  __syncthreads();
  float sq = 0.f;
#pragma unroll
  for (int i = 0; i < 4; ++i) { float d = v[i] - mu; sq += d * d; }
  red[tid] = sq; __syncthreads();
  for (int o = 128; o > 0; o >>= 1) { if (tid < o) red[tid] += red[tid + o]; __syncthreads(); }
  float rstd = rsqrtf(red[0] / D + 1e-5f);
#pragma unroll
  for (int i = 0; i < 4; ++i) {
    int c = tid + i * 256;
    float o = (v[i] - mu) * rstd * g[c] + bta[c];
    outf[(size_t)row * D + c] = o;
    if (outb) outb[(size_t)row * D + c] = f2bf(o);
  }
}

// SwiGLU gate: hg = silu(h[:, :2048]) * h[:, 2048:], zeroed on padded rows
__global__ void swiglu_kernel(const u16* __restrict__ h, const unsigned char* __restrict__ pad,
                              u16* __restrict__ hg)
{
  const int n = 4096 * 2048;
  for (int i = blockIdx.x * blockDim.x + threadIdx.x; i < n; i += gridDim.x * blockDim.x) {
    int row = i >> 11, c = i & 2047;
    float h1 = bf2f(h[(size_t)row * 4096 + c]);
    float h2 = bf2f(h[(size_t)row * 4096 + 2048 + c]);
    float v = pad[row] ? 0.f : (h1 / (1.f + __expf(-h1))) * h2;
    hg[i] = f2bf(v);
  }
}

__global__ void f32_to_bf16_kernel(const float* __restrict__ in, u16* __restrict__ out, int n)
{
  for (int i = blockIdx.x * blockDim.x + threadIdx.x; i < n; i += gridDim.x * blockDim.x)
    out[i] = f2bf(in[i]);
}

// ---------------------------------------------------------------------------
extern "C" void kernel_launch(void* const* d_in, const int* in_sizes, int n_in,
                              void* d_out, int out_size, void* d_ws, size_t ws_size,
                              hipStream_t stream)
{
  (void)in_sizes; (void)n_in; (void)out_size; (void)ws_size;
  constexpr int B = 2, S = 2048, D = 1024, H = 16, KV = 4, DK = 64, FF = 4096;
  constexpr int M = B * S;                     // 4096 rows
  constexpr int QKVN = D + 2 * KV * DK;        // 1536

  const float*         x     = (const float*)d_in[0];
  const int*           amask = (const int*)d_in[1];
  const unsigned char* smask = (const unsigned char*)d_in[2];   // bool array
  const float* Wc = (const float*)d_in[3];  const float* bc = (const float*)d_in[4];
  const float* Wo = (const float*)d_in[5];  const float* bo = (const float*)d_in[6];
  const float* W1 = (const float*)d_in[7];  const float* b1 = (const float*)d_in[8];
  const float* W2 = (const float*)d_in[9];  const float* b2 = (const float*)d_in[10];
  const float* g1 = (const float*)d_in[11]; const float* be1 = (const float*)d_in[12];
  const float* g2 = (const float*)d_in[13]; const float* be2 = (const float*)d_in[14];

  size_t off = 0;
  auto alloc = [&](size_t bytes) -> void* {
    void* p = (char*)d_ws + off;
    off += (bytes + 255) & ~(size_t)255;
    return p;
  };
  u16*   Xbf   = (u16*)  alloc((size_t)M * D * 2);
  u16*   Wcbf  = (u16*)  alloc((size_t)D * QKVN * 2);
  u16*   Wobf  = (u16*)  alloc((size_t)D * D * 2);
  u16*   W1bf  = (u16*)  alloc((size_t)D * FF * 2);
  u16*   W2bf  = (u16*)  alloc((size_t)(FF / 2) * D * 2);
  u16*   Qb    = (u16*)  alloc((size_t)B * H * S * DK * 2);
  u16*   Kb    = (u16*)  alloc((size_t)B * KV * S * DK * 2);
  u16*   Vb    = (u16*)  alloc((size_t)B * KV * S * DK * 2);
  u16*   ctxbf = (u16*)  alloc((size_t)M * D * 2);
  float* aout  = (float*)alloc((size_t)M * D * 4);
  float* x1f   = (float*)alloc((size_t)M * D * 4);
  u16*   x1bf  = (u16*)  alloc((size_t)M * D * 2);
  u16*   hbf   = (u16*)  alloc((size_t)M * FF * 2);
  u16*   hgbf  = (u16*)  alloc((size_t)M * (FF / 2) * 2);
  float* ffn   = (float*)alloc((size_t)M * D * 4);

  // 1) bf16 conversions (weights converted once per call; reused across 4096 rows)
  f32_to_bf16_kernel<<<2048, 256, 0, stream>>>(x,  Xbf,  M * D);
  f32_to_bf16_kernel<<<2048, 256, 0, stream>>>(Wc, Wcbf, D * QKVN);
  f32_to_bf16_kernel<<<2048, 256, 0, stream>>>(Wo, Wobf, D * D);
  f32_to_bf16_kernel<<<2048, 256, 0, stream>>>(W1, W1bf, D * FF);
  f32_to_bf16_kernel<<<2048, 256, 0, stream>>>(W2, W2bf, (FF / 2) * D);

  // 2) QKV projection (+bias, scatter to per-head bf16 Q/K/V)
  gemm_bf16<2><<<(M / 128) * (QKVN / 128), 256, 0, stream>>>(
      Xbf, Wcbf, bc, nullptr, nullptr, Qb, Kb, Vb, M, QKVN, D);

  // 3) Flash attention with ALiBi + mask
  attention_kernel<<<dim3(S / 64, H, B), 128, 0, stream>>>(Qb, Kb, Vb, amask, ctxbf);

  // 4) Output projection
  gemm_bf16<0><<<(M / 128) * (D / 128), 256, 0, stream>>>(
      ctxbf, Wobf, bo, aout, nullptr, nullptr, nullptr, nullptr, M, D, D);

  // 5) x1 = LN(x + attn_out)
  ln_kernel<<<M, 256, 0, stream>>>(x, aout, nullptr, g1, be1, x1f, x1bf);

  // 6) FFN up-projection (+b1), bf16 out
  gemm_bf16<1><<<(M / 128) * (FF / 128), 256, 0, stream>>>(
      x1bf, W1bf, b1, nullptr, hbf, nullptr, nullptr, nullptr, M, FF, D);

  // 7) SwiGLU gate (+pad mask)
  swiglu_kernel<<<2048, 256, 0, stream>>>(hbf, smask, hgbf);

  // 8) FFN down-projection (+b2)
  gemm_bf16<0><<<(M / 128) * (D / 128), 256, 0, stream>>>(
      hgbf, W2bf, b2, ffn, nullptr, nullptr, nullptr, nullptr, M, D, FF / 2);

  // 9) out = LN(x1 + masked(ffn))
  ln_kernel<<<M, 256, 0, stream>>>(x1f, ffn, smask, g2, be2, (float*)d_out, nullptr);
}
